// GraphSAGEWeight_11227044511906
// MI455X (gfx1250) — matrix operand order repaired and checked by reading, not compiled, not run on hardware
//
#include <hip/hip_runtime.h>
#include <hip/hip_bf16.h>
#include <math.h>

#define N_NODES   100000
#define N_EDGES   1600000
#define DIM       128
#define KP        136        // padded LDS row stride (bf16 elems), 16B-aligned rows
#define N_CLASSES 10
#define N_GRAPHS  128

typedef __attribute__((ext_vector_type(16))) __bf16 v16bf;
typedef __attribute__((ext_vector_type(8)))  __bf16 v8bf;
typedef __attribute__((ext_vector_type(8)))  float  v8f;
typedef __attribute__((ext_vector_type(4)))  unsigned int v4u;
typedef __attribute__((ext_vector_type(8)))  int    v8i;
typedef __attribute__((ext_vector_type(4)))  int    v4i;

// ---------------------------------------------------------------- utilities
__global__ void zero_f32_kernel(float* p, long long n) {
    long long i = (long long)blockIdx.x * blockDim.x + threadIdx.x;
    if (i < n) p[i] = 0.0f;
}

__global__ void degree_kernel(const long long* edge_index, float* deg, long long E) {
    long long e = (long long)blockIdx.x * blockDim.x + threadIdx.x;
    if (e >= E) return;
    long long d = edge_index[E + e];
    atomicAdd(&deg[d], 1.0f);
}

__global__ void invdeg_kernel(float* deg, int n) {
    int i = blockIdx.x * blockDim.x + threadIdx.x;
    if (i >= n) return;
    deg[i] = 1.0f / fmaxf(deg[i], 1.0f);
}

// agg[dst, f] += h[src, f] * ew[e]    (one thread per (edge, feature))
__global__ void scatter_kernel(const float* __restrict__ h,
                               const long long* __restrict__ edge_index,
                               const float* __restrict__ ew,
                               float* __restrict__ agg, long long E) {
    long long tid = (long long)blockIdx.x * blockDim.x + threadIdx.x;
    if (tid >= E * DIM) return;
    long long e = tid >> 7;
    int       f = (int)(tid & 127);
    long long s = edge_index[e];
    long long d = edge_index[E + e];
    float v = h[s * DIM + f] * ew[e];
    atomicAdd(&agg[d * DIM + f], v);
}

// ------------------------------------------------- TDM helper (ISA §8.3/8.4)
// One descriptor: 1-D tile of 16384 f32 elements, global -> LDS.
__device__ __forceinline__ void tdm_load_w(const float* gsrc, unsigned lds_off) {
    unsigned long long ga = (unsigned long long)gsrc;
    v4u g0;
    g0[0] = 1u;                                        // count=1, user D#
    g0[1] = lds_off;                                   // lds_addr (bytes)
    g0[2] = (unsigned)(ga & 0xFFFFFFFFu);              // global_addr[31:0]
    g0[3] = (unsigned)((ga >> 32) & 0x1FFFFFFu)        // global_addr[56:32]
          | (2u << 30);                                // type = 2 ("image")
    v8i g1;
    g1[0] = 0x00020000;                                // wg_mask=0, data_size=4B
    g1[1] = (int)((16384u & 0xFFFFu) << 16);           // tensor_dim0 lo16
    g1[2] = (int)((16384u >> 16) | (1u << 16));        // dim0 hi16 | tensor_dim1=1
    g1[3] = (int)(16384u << 16);                       // tile_dim0 = 16384
    g1[4] = 1;                                         // tile_dim1=1, tile_dim2=0
    g1[5] = 16384;                                     // tensor_dim0_stride
    g1[6] = 0;
    g1[7] = 0;
    v4i z4 = {0, 0, 0, 0};
    v8i z8 = {0, 0, 0, 0, 0, 0, 0, 0};
    __builtin_amdgcn_tensor_load_to_lds(g0, g1, z4, z4, z8, 0);
}

// -------------------------------------------------- fused SAGE linear (WMMA)
// out = relu( (agg * inv_deg) @ Wl + h @ Wr + bias )
// Block = 128 threads = 4 waves; each wave owns one 16-row strip of A.
// Weights staged by the Tensor Data Mover into LDS (f32), then converted to
// a transposed, padded bf16 layout so each B fragment is 2x ds_load_b128.
__global__ void sage_gemm_kernel(const float* __restrict__ agg,
                                 const float* __restrict__ h,
                                 const float* __restrict__ inv_deg,
                                 const float* __restrict__ Wl,
                                 const float* __restrict__ Wr,
                                 const float* __restrict__ bias,
                                 float* __restrict__ out, int nrows) {
    __shared__ float  sStage[DIM * DIM];        // 64 KB f32 staging (TDM dest)
    __shared__ __bf16 sB[2 * DIM * KP];         // 68 KB bf16, transposed+padded

    const int tid = threadIdx.x;

    // stage + convert each weight matrix: TDM load -> wait -> transpose/cvt
    for (int mat = 0; mat < 2; ++mat) {
        const float* src = (mat == 0) ? Wl : Wr;
        if ((tid >> 5) == 0) {                  // wave 0 drives the TDM
            tdm_load_w(src, (unsigned)(unsigned long long)&sStage[0]);
            __builtin_amdgcn_s_wait_tensorcnt(0);
        }
        __syncthreads();
        for (int i = tid; i < DIM * DIM; i += 128) {
            int k = i >> 7, n = i & 127;        // W[k][n] -> sB[mat][n][k]
            sB[mat * DIM * KP + n * KP + k] = (__bf16)sStage[i];
        }
        __syncthreads();
    }

    const int wave  = tid >> 5;
    const int lane  = tid & 31;
    const int grp   = lane >> 4;                // lane group (K-half select)
    const int m     = lane & 15;
    const int strip = blockIdx.x * 4 + wave;
    const int row0  = strip * 16;
    if (row0 >= nrows) return;

    const int row   = row0 + m;
    const float scl = inv_deg[row];
    const float* arow = agg + (long long)row * DIM;
    const float* hrow = h   + (long long)row * DIM;

    v8f acc[8];
    for (int t = 0; t < 8; ++t) {
        float bv = bias[t * 16 + m];
        for (int r = 0; r < 8; ++r) acc[t][r] = bv;
    }

    for (int kk = 0; kk < 4; ++kk) {            // K steps of 32
        const int kbase = kk * 32;
        if (kk < 3) {
            __builtin_prefetch(arow + kbase + 32, 0, 1);
            __builtin_prefetch(hrow + kbase + 32, 0, 1);
        }
        // A fragment: two contiguous 8-float runs per lane (ISA 16b A layout)
        v8f c1 = *(const v8f*)(arow + kbase + grp * 8);
        v8f c2 = *(const v8f*)(arow + kbase + 16 + grp * 8);
        v8f d1 = *(const v8f*)(hrow + kbase + grp * 8);
        v8f d2 = *(const v8f*)(hrow + kbase + 16 + grp * 8);
        v16bf a1, a2;
        #pragma unroll
        for (int i = 0; i < 8; ++i) {
            a1[i]     = (__bf16)(c1[i] * scl);
            a1[8 + i] = (__bf16)(c2[i] * scl);
            a2[i]     = (__bf16)d1[i];
            a2[8 + i] = (__bf16)d2[i];
        }
        #pragma unroll
        for (int t = 0; t < 8; ++t) {
            const int n   = t * 16 + m;
            const int off = n * KP + kbase + grp * 8;    // 16B-aligned
            v8bf bl0 = *(const v8bf*)&sB[off];
            v8bf bl1 = *(const v8bf*)&sB[off + 16];
            v8bf br0 = *(const v8bf*)&sB[DIM * KP + off];
            v8bf br1 = *(const v8bf*)&sB[DIM * KP + off + 16];
            v16bf bl = __builtin_shufflevector(bl0, bl1, 0, 1, 2, 3, 4, 5, 6, 7,
                                               8, 9, 10, 11, 12, 13, 14, 15);
            v16bf br = __builtin_shufflevector(br0, br1, 0, 1, 2, 3, 4, 5, 6, 7,
                                               8, 9, 10, 11, 12, 13, 14, 15);
            acc[t] = __builtin_amdgcn_wmma_f32_16x16x32_bf16(
                false, a1, false, bl, (short)0, acc[t], false, false);
            acc[t] = __builtin_amdgcn_wmma_f32_16x16x32_bf16(
                false, a2, false, br, (short)0, acc[t], false, false);
        }
    }

    // C/D layout: VGPR r, lanes 0-15 -> M=r, lanes 16-31 -> M=r+8, N=lane%16
    #pragma unroll
    for (int t = 0; t < 8; ++t) {
        #pragma unroll
        for (int r = 0; r < 8; ++r) {
            int gr = row0 + grp * 8 + r;
            out[(long long)gr * DIM + t * 16 + m] = fmaxf(acc[t][r], 0.0f);
        }
    }
}

// ------------------------------------------------------------------ pooling
__global__ void pool_kernel(const float* __restrict__ h,
                            const long long* __restrict__ batch,
                            float* __restrict__ pooled, float* __restrict__ cnt,
                            long long n) {
    long long tid = (long long)blockIdx.x * blockDim.x + threadIdx.x;
    if (tid >= n * DIM) return;
    long long i = tid >> 7;
    int       f = (int)(tid & 127);
    long long g = batch[i];
    atomicAdd(&pooled[g * DIM + f], h[i * DIM + f]);
    if (f == 0) atomicAdd(&cnt[g], 1.0f);
}

// ------------------------------------------------------- classifier head
__global__ void head_kernel(const float* __restrict__ pooled,
                            const float* __restrict__ cnt,
                            const float* __restrict__ w1, const float* __restrict__ b1,
                            const float* __restrict__ w2, const float* __restrict__ b2,
                            float* __restrict__ out) {
    __shared__ float z[DIM];
    __shared__ float logit[16];
    const int g = blockIdx.x;
    const int n = threadIdx.x;
    const float inv = 1.0f / fmaxf(cnt[g], 1.0f);
    float acc = b1[n];
    for (int k = 0; k < DIM; ++k)
        acc += pooled[g * DIM + k] * inv * w1[k * DIM + n];
    z[n] = fmaxf(acc, 0.0f);
    __syncthreads();
    if (n < N_CLASSES) {
        float a = b2[n];
        for (int k = 0; k < DIM; ++k) a += z[k] * w2[k * N_CLASSES + n];
        logit[n] = a;
    }
    __syncthreads();
    if (n == 0) {
        float mx = -INFINITY;
        for (int c = 0; c < N_CLASSES; ++c) mx = fmaxf(mx, logit[c]);
        float s = 0.0f;
        for (int c = 0; c < N_CLASSES; ++c) s += expf(logit[c] - mx);
        float lse = mx + logf(s);
        for (int c = 0; c < N_CLASSES; ++c) out[g * N_CLASSES + c] = logit[c] - lse;
    }
}

// ------------------------------------------------------------------ launch
extern "C" void kernel_launch(void* const* d_in, const int* in_sizes, int n_in,
                              void* d_out, int out_size, void* d_ws, size_t ws_size,
                              hipStream_t stream) {
    const float*     x      = (const float*)d_in[0];
    const float*     ew     = (const float*)d_in[1];
    const long long* eidx   = (const long long*)d_in[2];
    const long long* batch  = (const long long*)d_in[3];
    const float*     W1_l   = (const float*)d_in[4];
    const float*     W1_r   = (const float*)d_in[5];
    const float*     b1     = (const float*)d_in[6];
    const float*     Ws_l   = (const float*)d_in[7];
    const float*     Ws_r   = (const float*)d_in[8];
    const float*     bs     = (const float*)d_in[9];
    const float*     lin1_w = (const float*)d_in[10];
    const float*     lin1_b = (const float*)d_in[11];
    const float*     lin2_w = (const float*)d_in[12];
    const float*     lin2_b = (const float*)d_in[13];
    float* out = (float*)d_out;

    float* ws      = (float*)d_ws;
    const long long NF = (long long)N_NODES * DIM;
    float* agg     = ws;
    float* hA      = agg + NF;
    float* hB      = hA + NF;
    float* invdeg  = hB + NF;
    float* pooled  = invdeg + N_NODES;
    float* cnt     = pooled + N_GRAPHS * DIM;

    const long long E = N_EDGES;
    const int TPB = 256;
    dim3 blk(TPB);
    long long escat = E * DIM;
    dim3 gScat((unsigned)((escat + TPB - 1) / TPB));
    dim3 gZeroNF((unsigned)((NF + TPB - 1) / TPB));
    int nstrips = N_NODES / 16;                 // 6250
    dim3 gGemm((unsigned)((nstrips + 3) / 4));
    dim3 bGemm(128);

    zero_f32_kernel<<<dim3((N_NODES + TPB - 1) / TPB), blk, 0, stream>>>(invdeg, N_NODES);
    degree_kernel<<<dim3((unsigned)((E + TPB - 1) / TPB)), blk, 0, stream>>>(eidx, invdeg, E);
    invdeg_kernel<<<dim3((N_NODES + TPB - 1) / TPB), blk, 0, stream>>>(invdeg, N_NODES);

    // layer 1: x -> hA
    zero_f32_kernel<<<gZeroNF, blk, 0, stream>>>(agg, NF);
    scatter_kernel<<<gScat, blk, 0, stream>>>(x, eidx, ew, agg, E);
    sage_gemm_kernel<<<gGemm, bGemm, 0, stream>>>(agg, x, invdeg, W1_l, W1_r, b1, hA, N_NODES);

    // layer 2: hA -> hB
    zero_f32_kernel<<<gZeroNF, blk, 0, stream>>>(agg, NF);
    scatter_kernel<<<gScat, blk, 0, stream>>>(hA, eidx, ew, agg, E);
    sage_gemm_kernel<<<gGemm, bGemm, 0, stream>>>(agg, hA, invdeg,
                                                  Ws_l, Ws_r, bs, hB, N_NODES);

    // layer 3: hB -> hA
    zero_f32_kernel<<<gZeroNF, blk, 0, stream>>>(agg, NF);
    scatter_kernel<<<gScat, blk, 0, stream>>>(hB, eidx, ew, agg, E);
    sage_gemm_kernel<<<gGemm, bGemm, 0, stream>>>(agg, hB, invdeg,
                                                  Ws_l + DIM * DIM, Ws_r + DIM * DIM,
                                                  bs + DIM, hA, N_NODES);

    // global mean pool + head
    zero_f32_kernel<<<dim3((N_GRAPHS * DIM + N_GRAPHS + TPB - 1) / TPB), blk, 0, stream>>>(
        pooled, N_GRAPHS * DIM + N_GRAPHS);
    pool_kernel<<<dim3((unsigned)(((long long)N_NODES * DIM + TPB - 1) / TPB)), blk, 0, stream>>>(
        hA, batch, pooled, cnt, N_NODES);
    head_kernel<<<dim3(N_GRAPHS), dim3(DIM), 0, stream>>>(
        pooled, cnt, lin1_w, lin1_b, lin2_w, lin2_b, out);
}
